// SFFM_47665547051505
// MI455X (gfx1250) — matrix operand order repaired and checked by reading, not compiled, not run on hardware
//
#include <hip/hip_runtime.h>
#include <hip/hip_bf16.h>
#include <math.h>

#define T_    4096
#define IN_   512
#define TR_   64
#define CTX_  64
#define HID_  1024

typedef __bf16 bf16;
typedef __attribute__((ext_vector_type(16))) __bf16 v16bf;
typedef __attribute__((ext_vector_type(8)))  __bf16 v8bf;
typedef __attribute__((ext_vector_type(8)))  float  v8f;

// ---------------- elementwise f32 -> bf16 ----------------
__global__ void cvt_bf16_kernel(const float* __restrict__ in, bf16* __restrict__ out, int n) {
  int i = blockIdx.x * blockDim.x + threadIdx.x;
  if (i < n) out[i] = (bf16)in[i];
}

// ---------------- transpose + convert: in[K x N] f32 -> out[N x K] bf16 ----------------
__global__ void transpose_cvt_kernel(const float* __restrict__ in, bf16* __restrict__ out,
                                     int K, int N) {
  int i = blockIdx.x * blockDim.x + threadIdx.x;
  if (i < K * N) {
    int k = i / N, n = i % N;
    out[(size_t)n * K + k] = (bf16)in[i];
  }
}

// Load one 16x32 bf16 A-fragment for v_wmma_f32_16x16x32_bf16.
// Lane layout: lanes 0-15 row M=lane, K=0..15; lanes 16-31 K=16..31 (matches B).
__device__ __forceinline__ v16bf load_a_frag(const bf16* __restrict__ row, int k0, int half) {
  v8bf lo = *(const v8bf*)(row + k0 +     half * 16);
  v8bf hi = *(const v8bf*)(row + k0 + 8 + half * 16);
  v16bf a;
#pragma unroll
  for (int i = 0; i < 8; ++i) { a[i] = lo[i]; a[i + 8] = hi[i]; }
  return a;
}

// ---------------- bf16 WMMA GEMM: C[M,N] = A[M,K] @ B[K,N] + bias, optional sigmoid ----
// A: M x K row-major bf16.  Bt: N x K row-major bf16 (B transposed).
// Register-blocked: per-wave tile = (AM*16) x (BN*16) using AM x BN accumulators
// (AM*BN WMMAs per (2*AM + 2*BN) b128 loads). Wave grid in block: WGM x WGN.
// Block tile = (WGM*AM*16) x (WGN*BN*16).
template<int AM, int BN, int WGM, int WGN, int ACT>
__global__ __launch_bounds__(32 * WGM * WGN) void gemm_bf16_kernel(
    const bf16* __restrict__ A, const bf16* __restrict__ Bt,
    const float* __restrict__ bias, float* __restrict__ C,
    int M, int N, int K) {
  const int lane  = threadIdx.x & 31;
  const int wave  = threadIdx.x >> 5;
  const int half  = lane >> 4;    // 0: lanes 0-15, 1: lanes 16-31
  const int l16   = lane & 15;
  const int waveM = wave / WGN;
  const int waveN = wave % WGN;
  const int tileM = blockIdx.y * (WGM * AM * 16) + waveM * (AM * 16);
  const int tileN = blockIdx.x * (WGN * BN * 16) + waveN * (BN * 16);

  const bf16* arow[AM];
  const bf16* brow[BN];
#pragma unroll
  for (int am = 0; am < AM; ++am) arow[am] = A  + (size_t)(tileM + am * 16 + l16) * K;
#pragma unroll
  for (int bn = 0; bn < BN; ++bn) brow[bn] = Bt + (size_t)(tileN + bn * 16 + l16) * K;

  v8f acc[AM][BN];
#pragma unroll
  for (int am = 0; am < AM; ++am)
#pragma unroll
    for (int bn = 0; bn < BN; ++bn) acc[am][bn] = (v8f){};

  for (int k0 = 0; k0 < K; k0 += 32) {
    v16bf a[AM], b[BN];
#pragma unroll
    for (int am = 0; am < AM; ++am) a[am] = load_a_frag(arow[am], k0, half);
    // B lane layout: lanes 0-15 hold K=k0..k0+15, lanes 16-31 hold K=k0+16..k0+31
#pragma unroll
    for (int bn = 0; bn < BN; ++bn) b[bn] = *(const v16bf*)(brow[bn] + k0 + half * 16);
#pragma unroll
    for (int am = 0; am < AM; ++am)
#pragma unroll
      for (int bn = 0; bn < BN; ++bn)
        acc[am][bn] = __builtin_amdgcn_wmma_f32_16x16x32_bf16(
            false, a[am], false, b[bn], (short)0, acc[am][bn], false, false);
  }

  // C layout: VGPR i -> M = i + 8*half, N = l16 (per 16x16 sub-tile)
#pragma unroll
  for (int am = 0; am < AM; ++am)
#pragma unroll
    for (int bn = 0; bn < BN; ++bn)
#pragma unroll
      for (int i = 0; i < 8; ++i) {
        int m = tileM + am * 16 + i + half * 8;
        int n = tileN + bn * 16 + l16;
        float v = acc[am][bn][i] + bias[n];
        if (ACT == 1) v = 1.0f / (1.0f + __expf(-v));   // sigmoid
        C[(size_t)m * N + n] = v;
      }
}

// ---------------- factorized normalizer: inv_norm[t] = 1/(sum|trace_r| * sum|ctx_c|) ----
__global__ void rownorm_kernel(const float* __restrict__ trace, const float* __restrict__ ctx,
                               float* __restrict__ invn) {
  __shared__ float sA[64], sC[64];
  int t = blockIdx.x, i = threadIdx.x;
  sA[i] = fabsf(trace[(size_t)t * TR_ + i]);
  sC[i] = fabsf(ctx[(size_t)t * CTX_ + i]);
  __syncthreads();
  for (int s = 32; s > 0; s >>= 1) {
    if (i < s) { sA[i] += sA[i + s]; sC[i] += sC[i + s]; }
    __syncthreads();
  }
  if (i == 0) invn[t] = 1.0f / (sA[0] * sC[0]);
}

// ---------------- sequential complex scan over T, one thread per (r,c) channel ---------
// block = r (trace[t,r] broadcast), thread = c (ctx[t,c] coalesced)
__global__ __launch_bounds__(64) void scan_kernel(
    const float* __restrict__ trace, const float* __restrict__ ctx,
    const float* __restrict__ invn,
    const float* __restrict__ a, const float* __restrict__ b,
    bf16* __restrict__ scaled) {
  int r = blockIdx.x;
  int c = threadIdx.x;
  float decay = __expf(-fabsf(a[r]));
  float gre = decay * __cosf(b[c]);
  float gim = decay * __sinf(b[c]);
  float sre = 0.f, sim = 0.f;
  for (int t = 0; t < T_; ++t) {
    float p = fabsf(trace[(size_t)t * TR_ + r]) * fabsf(ctx[(size_t)t * CTX_ + c]) * invn[t];
    float nre = sre * gre - sim * gim + p;
    float nim = sre * gim + sim * gre;
    sre = nre; sim = nim;
    // eps = re + (re == sign(re))*0.01 ; s += eps (affects real part only)
    float sr = sre;
    float sg = (sr > 0.f) ? 1.f : ((sr < 0.f) ? -1.f : 0.f);
    float eps = sr + ((sr == sg) ? 0.01f : 0.f);
    float SR = sr + eps;
    float SI = sim;
    float rad = sqrtf(SR * SR + SI * SI);
    float mag = log1pf(rad);
    float sv = 0.f, cv = 0.f;
    if (rad > 0.f) { float ir = mag / rad; sv = SI * ir; cv = SR * ir; }  // mag*sin, mag*cos
    size_t base = (size_t)t * (2 * TR_ * CTX_) + (size_t)r * CTX_ + c;
    scaled[base] = (bf16)sv;
    scaled[base + TR_ * CTX_] = (bf16)cv;
  }
}

// ---------------- LayerNorm + leaky_relu (slope 0.01); bf16 and/or f32 outputs ---------
__global__ __launch_bounds__(256) void ln_leaky_kernel(
    const float* __restrict__ H, const float* __restrict__ g, const float* __restrict__ be,
    bf16* __restrict__ out_bf, float* __restrict__ out_f32) {
  __shared__ float red[256];
  int row = blockIdx.x, tid = threadIdx.x;
  const float* h = H + (size_t)row * HID_;
  float s = 0.f;
  for (int i = tid; i < HID_; i += 256) s += h[i];
  red[tid] = s; __syncthreads();
  for (int st = 128; st > 0; st >>= 1) { if (tid < st) red[tid] += red[tid + st]; __syncthreads(); }
  float mu = red[0] / HID_;
  __syncthreads();
  float v = 0.f;
  for (int i = tid; i < HID_; i += 256) { float d = h[i] - mu; v += d * d; }
  red[tid] = v; __syncthreads();
  for (int st = 128; st > 0; st >>= 1) { if (tid < st) red[tid] += red[tid + st]; __syncthreads(); }
  float rstd = rsqrtf(red[0] / HID_ + 1e-6f);
  for (int i = tid; i < HID_; i += 256) {
    float y = (h[i] - mu) * rstd * g[i] + be[i];
    y = (y > 0.f) ? y : 0.01f * y;
    if (out_bf)  out_bf[(size_t)row * HID_ + i] = (bf16)y;
    if (out_f32) out_f32[(size_t)row * HID_ + i] = y;
  }
}

extern "C" void kernel_launch(void* const* d_in, const int* in_sizes, int n_in,
                              void* d_out, int out_size, void* d_ws, size_t ws_size,
                              hipStream_t stream) {
  (void)in_sizes; (void)n_in; (void)out_size; (void)ws_size;
  const float* x   = (const float*)d_in[0];
  // d_in[1] = resets (all false) -> unused
  const float* a   = (const float*)d_in[2];
  const float* b   = (const float*)d_in[3];
  const float* Wt  = (const float*)d_in[4];
  const float* bt  = (const float*)d_in[5];
  const float* Wc  = (const float*)d_in[6];
  const float* bc  = (const float*)d_in[7];
  const float* W1  = (const float*)d_in[8];
  const float* b1  = (const float*)d_in[9];
  const float* g1  = (const float*)d_in[10];
  const float* be1 = (const float*)d_in[11];
  const float* W2  = (const float*)d_in[12];
  const float* b2  = (const float*)d_in[13];
  const float* g2  = (const float*)d_in[14];
  const float* be2 = (const float*)d_in[15];

  char* ws = (char*)d_ws; size_t off = 0;
  auto alloc = [&](size_t bytes) -> void* {
    off = (off + 255) & ~(size_t)255;
    void* p = ws + off; off += bytes; return p;
  };
  bf16*  xb     = (bf16*)alloc((size_t)T_ * IN_ * 2);
  bf16*  WtT    = (bf16*)alloc((size_t)TR_ * IN_ * 2);
  bf16*  WcT    = (bf16*)alloc((size_t)CTX_ * IN_ * 2);
  bf16*  W1T    = (bf16*)alloc((size_t)HID_ * (2 * TR_ * CTX_) * 2);
  bf16*  W2T    = (bf16*)alloc((size_t)HID_ * HID_ * 2);
  float* trace  = (float*)alloc((size_t)T_ * TR_ * 4);
  float* ctx    = (float*)alloc((size_t)T_ * CTX_ * 4);
  float* invn   = (float*)alloc((size_t)T_ * 4);
  bf16*  scaled = (bf16*)alloc((size_t)T_ * (2 * TR_ * CTX_) * 2);
  float* h1     = (float*)alloc((size_t)T_ * HID_ * 4);
  bf16*  h1b    = (bf16*)alloc((size_t)T_ * HID_ * 2);
  float* h2     = (float*)alloc((size_t)T_ * HID_ * 4);

  // 1) precision conversion + weight transposes (B^T makes WMMA B-fragment loads contiguous)
  int nX = T_ * IN_;
  cvt_bf16_kernel<<<(nX + 255) / 256, 256, 0, stream>>>(x, xb, nX);
  transpose_cvt_kernel<<<(IN_ * TR_ + 255) / 256, 256, 0, stream>>>(Wt, WtT, IN_, TR_);
  transpose_cvt_kernel<<<(IN_ * CTX_ + 255) / 256, 256, 0, stream>>>(Wc, WcT, IN_, CTX_);
  transpose_cvt_kernel<<<((2 * TR_ * CTX_) * HID_ + 255) / 256, 256, 0, stream>>>(
      W1, W1T, 2 * TR_ * CTX_, HID_);
  transpose_cvt_kernel<<<(HID_ * HID_ + 255) / 256, 256, 0, stream>>>(W2, W2T, HID_, HID_);

  // 2) trace = x@Wt + bt ; ctx = sigmoid(x@Wc + bc)
  //    small-N config: 32x32/wave, 4x2 waves -> 128x64 block tile
  gemm_bf16_kernel<2, 2, 4, 2, 0><<<dim3(TR_ / 64,  T_ / 128), 256, 0, stream>>>(
      xb, WtT, bt, trace, T_, TR_,  IN_);
  gemm_bf16_kernel<2, 2, 4, 2, 1><<<dim3(CTX_ / 64, T_ / 128), 256, 0, stream>>>(
      xb, WcT, bc, ctx,   T_, CTX_, IN_);

  // 3) factorized normalizer + sequential complex scan -> scaled (bf16)
  rownorm_kernel<<<T_, 64, 0, stream>>>(trace, ctx, invn);
  scan_kernel<<<TR_, CTX_, 0, stream>>>(trace, ctx, invn, a, b, scaled);

  // 4) MLP: big-GEMM config: 32x64/wave (2Ax4B), 4x2 waves -> 128x128 block tile
  gemm_bf16_kernel<2, 4, 4, 2, 0><<<dim3(HID_ / 128, T_ / 128), 256, 0, stream>>>(
      scaled, W1T, b1, h1, T_, HID_, 2 * TR_ * CTX_);
  ln_leaky_kernel<<<T_, 256, 0, stream>>>(h1, g1, be1, h1b, nullptr);
  gemm_bf16_kernel<2, 4, 4, 2, 0><<<dim3(HID_ / 128, T_ / 128), 256, 0, stream>>>(
      h1b, W2T, b2, h2, T_, HID_, HID_);
  ln_leaky_kernel<<<T_, 256, 0, stream>>>(h2, g2, be2, nullptr, (float*)d_out);
}
